// NoisyTopKRouter_24859270709998
// MI455X (gfx1250) — matrix operand order repaired and checked by reading, not compiled, not run on hardware
//
#include <hip/hip_runtime.h>
#include <math.h>

typedef __bf16 bf16_t;
typedef __bf16 v16bf __attribute__((ext_vector_type(16)));
typedef float  v8f   __attribute__((ext_vector_type(8)));

#define C_DIM   2048
#define E_DIM   64
#define N_TOK   16384          // B*T = 4*4096
#define TOPK    8
#define KCHUNK  32
#define NCHUNK  (C_DIM / KCHUNK)
#define WSTRIDE 40             // padded bf16 row stride (80B, 16B aligned)
#define WBUFSZ  (4 * E_DIM * WSTRIDE)   // bf16 elements per staging buffer
#define LSTRIDE 65             // padded float row stride (bank-conflict-free)

// B-fragment loader: slot = {0:route_hi, 1:route_lo, 2:noise_hi, 3:noise_lo}
static __device__ __forceinline__ v16bf load_b_frag(const bf16_t* wb, int slot,
                                                    int tile, int mrow, int kloB) {
    const bf16_t* p = wb + ((size_t)(slot * E_DIM + tile * 16 + mrow)) * WSTRIDE + kloB;
    v16bf b;
    ((uint4*)&b)[0] = *(const uint4*)(p);
    ((uint4*)&b)[1] = *(const uint4*)(p + 16);
    return b;
}

// Cooperative: load one K-chunk of both weight matrices, split fp32 -> bf16 hi/lo,
// store into the given LDS staging buffer.
static __device__ __forceinline__ void stage_weights(bf16_t* wb,
                                                     const float* __restrict__ w_route,
                                                     const float* __restrict__ w_noise,
                                                     int kb, int tid) {
    // 2 mats * 64 experts * 8 float4-groups = 1024 groups
    for (int i = tid; i < 1024; i += 128) {
        int m  = i >> 9;            // 0: route, 1: noise
        int r  = i & 511;
        int e  = r >> 3;            // expert 0..63
        int k4 = (r & 7) * 4;       // 0,4,...,28
        const float* src = (m ? w_noise : w_route) + (size_t)e * C_DIM + kb + k4;
        float4 v = *(const float4*)src;
        union { bf16_t b[4]; uint2 u; } H, L;
        H.b[0] = (bf16_t)v.x; L.b[0] = (bf16_t)(v.x - (float)H.b[0]);
        H.b[1] = (bf16_t)v.y; L.b[1] = (bf16_t)(v.y - (float)H.b[1]);
        H.b[2] = (bf16_t)v.z; L.b[2] = (bf16_t)(v.z - (float)H.b[2]);
        H.b[3] = (bf16_t)v.w; L.b[3] = (bf16_t)(v.w - (float)H.b[3]);
        *(uint2*)(wb + ((size_t)((m * 2 + 0) * E_DIM + e)) * WSTRIDE + k4) = H.u;
        *(uint2*)(wb + ((size_t)((m * 2 + 1) * E_DIM + e)) * WSTRIDE + k4) = L.u;
    }
}

__global__ __launch_bounds__(128)
void noisy_topk_router_kernel(const float* __restrict__ x,
                              const float* __restrict__ w_route,
                              const float* __restrict__ w_noise,
                              const float* __restrict__ noise,
                              float* __restrict__ out)
{
    __shared__ __align__(16) bf16_t s_wA[WBUFSZ];               // 20480 B
    __shared__ __align__(16) bf16_t s_wB[WBUFSZ];               // 20480 B
    __shared__ float s_lr[E_DIM][LSTRIDE];                      // route logits (later: noisy)
    __shared__ float s_ln[E_DIM][LSTRIDE];                      // noise logits

    const int tid  = threadIdx.x;
    const int lane = tid & 31;
    const int wid  = tid >> 5;                 // 0..3 (wave id)
    const int tok0 = blockIdx.x * 64;          // 64 tokens per block

    const int mrow = lane & 15;
    const int kloA = (lane < 16) ? 0 : 8;      // A-matrix 16-bit striping
    const int kloB = (lane < 16) ? 0 : 8;      // B-matrix 16-bit striping

    const float* xrow = x + (size_t)(tok0 + wid * 16 + mrow) * C_DIM;

    v8f accR[4], accN[4];
    v8f zero = {0.f, 0.f, 0.f, 0.f, 0.f, 0.f, 0.f, 0.f};
    #pragma unroll
    for (int t = 0; t < 4; ++t) { accR[t] = zero; accN[t] = zero; }

    // ---- prologue: stage chunk 0 ----
    bf16_t* curbuf = s_wA;
    bf16_t* nxtbuf = s_wB;
    stage_weights(curbuf, w_route, w_noise, 0, tid);
    __syncthreads();

    for (int c = 0; c < NCHUNK; ++c) {
        const int kb = c * KCHUNK;

        // ---- stage NEXT chunk into the other buffer (overlaps WMMAs below) ----
        if (c + 1 < NCHUNK)
            stage_weights(nxtbuf, w_route, w_noise, kb + KCHUNK, tid);

        // ---- prefetch x two chunks ahead ----
        if (kb + 2 * KCHUNK < C_DIM)
            __builtin_prefetch(xrow + kb + 2 * KCHUNK, 0, 0);

        // ---- A fragment: 16 bf16 per lane, hi/lo split ----
        const float* ap = xrow + kb + kloA;
        float4 a0 = *(const float4*)(ap + 0);
        float4 a1 = *(const float4*)(ap + 4);
        float4 a2 = *(const float4*)(ap + 16);
        float4 a3 = *(const float4*)(ap + 20);
        float av[16] = { a0.x, a0.y, a0.z, a0.w,  a1.x, a1.y, a1.z, a1.w,
                         a2.x, a2.y, a2.z, a2.w,  a3.x, a3.y, a3.z, a3.w };
        v16bf ah, al;
        #pragma unroll
        for (int i = 0; i < 16; ++i) {
            bf16_t h = (bf16_t)av[i];
            ah[i] = h;
            al[i] = (bf16_t)(av[i] - (float)h);
        }

        // ---- 2 mats x 4 expert tiles x 3 WMMA (bf16x3 fp32-recovery) ----
        #pragma unroll
        for (int t = 0; t < 4; ++t) {
            v16bf bh = load_b_frag(curbuf, 0, t, mrow, kloB);
            v16bf bl = load_b_frag(curbuf, 1, t, mrow, kloB);
            accR[t] = __builtin_amdgcn_wmma_f32_16x16x32_bf16(false, ah, false, bh, (short)0, accR[t], false, false);
            accR[t] = __builtin_amdgcn_wmma_f32_16x16x32_bf16(false, ah, false, bl, (short)0, accR[t], false, false);
            accR[t] = __builtin_amdgcn_wmma_f32_16x16x32_bf16(false, al, false, bh, (short)0, accR[t], false, false);

            v16bf nh = load_b_frag(curbuf, 2, t, mrow, kloB);
            v16bf nl = load_b_frag(curbuf, 3, t, mrow, kloB);
            accN[t] = __builtin_amdgcn_wmma_f32_16x16x32_bf16(false, ah, false, nh, (short)0, accN[t], false, false);
            accN[t] = __builtin_amdgcn_wmma_f32_16x16x32_bf16(false, ah, false, nl, (short)0, accN[t], false, false);
            accN[t] = __builtin_amdgcn_wmma_f32_16x16x32_bf16(false, al, false, nh, (short)0, accN[t], false, false);
        }

        // Single barrier per chunk: guarantees (a) all waves done reading curbuf
        // before it is overwritten next iteration, and (b) nxtbuf stores visible.
        __syncthreads();

        bf16_t* tmp = curbuf; curbuf = nxtbuf; nxtbuf = tmp;
    }

    // ---- scatter logits to LDS (C/D layout: row M = r + 8*(lane>=16), col = lane&15)
    {
        const int rbase = wid * 16 + ((lane < 16) ? 0 : 8);
        #pragma unroll
        for (int t = 0; t < 4; ++t) {
            #pragma unroll
            for (int r = 0; r < 8; ++r) {
                s_lr[rbase + r][t * 16 + mrow] = accR[t][r];
                s_ln[rbase + r][t * 16 + mrow] = accN[t][r];
            }
        }
    }
    __syncthreads();

    // ---- per-token: noisy logits, top-8, sparse softmax, outputs ----
    if (tid < 64) {
        const int token = tok0 + tid;
        const float* nrow = noise + (size_t)token * E_DIM;

        // noisy = logits + noise * softplus(noise_logits); overwrite s_lr row
        for (int e = 0; e < E_DIM; ++e) {
            float nl = s_ln[tid][e];
            float sp = (nl > 20.0f) ? nl : log1pf(expf(nl));
            s_lr[tid][e] = s_lr[tid][e] + nrow[e] * sp;
        }

        // top-8: ascending scan with strict '>' matches jax.lax.top_k tie-break
        int   idx[TOPK];
        float val[TOPK];
        unsigned long long taken = 0ull;
        for (int k = 0; k < TOPK; ++k) {
            float best = -__builtin_inff();
            int   bi   = 0;
            for (int e = 0; e < E_DIM; ++e) {
                if (!((taken >> e) & 1ull)) {
                    float v = s_lr[tid][e];
                    if (v > best) { best = v; bi = e; }
                }
            }
            taken |= (1ull << bi);
            idx[k] = bi;
            val[k] = best;
        }

        // softmax over the selected 8 (others are exp(-inf)=0)
        float mx = val[0];
        float wv[TOPK];
        float wsum = 0.f;
        #pragma unroll
        for (int k = 0; k < TOPK; ++k) { wv[k] = expf(val[k] - mx); wsum += wv[k]; }
        float inv = 1.0f / wsum;

        float* orow = out + (size_t)token * E_DIM;
        for (int e = 0; e < E_DIM; ++e) orow[e] = 0.0f;
        #pragma unroll
        for (int k = 0; k < TOPK; ++k) orow[idx[k]] = wv[k] * inv;

        float* irow = out + (size_t)N_TOK * E_DIM + (size_t)token * TOPK;
        #pragma unroll
        for (int k = 0; k < TOPK; ++k) irow[k] = (float)idx[k];
    }
}

extern "C" void kernel_launch(void* const* d_in, const int* in_sizes, int n_in,
                              void* d_out, int out_size, void* d_ws, size_t ws_size,
                              hipStream_t stream) {
    (void)in_sizes; (void)n_in; (void)out_size; (void)d_ws; (void)ws_size;
    const float* x  = (const float*)d_in[0];
    const float* wr = (const float*)d_in[1];
    const float* wn = (const float*)d_in[2];
    const float* nz = (const float*)d_in[3];
    float* out = (float*)d_out;
    noisy_topk_router_kernel<<<dim3(N_TOK / 64), dim3(128), 0, stream>>>(x, wr, wn, nz, out);
}
